// E74AblationCell_23751169147419
// MI455X (gfx1250) — matrix operand order repaired and checked by reading, not compiled, not run on hardware
//
#include <hip/hip_runtime.h>
#include <hip/hip_bf16.h>
#include <cstdint>
#include <cstddef>

// ---------------------------------------------------------------------------
// Problem constants (fixed by the reference)
// ---------------------------------------------------------------------------
#define TT      1024          // timesteps
#define BB      32            // batch
#define DD      1024          // input dim
#define NN      1024          // n_state
#define BLK     8             // block size
#define MROWS   (TT * BB)     // 32768 GEMM rows
#define BNFLAT  (BB * NN)     // 32768 elements per timestep plane

typedef __attribute__((ext_vector_type(16))) __bf16 v16bf;
typedef __attribute__((ext_vector_type(8)))  __bf16 v8bf;
typedef __attribute__((ext_vector_type(8)))  float  v8f;

// ---------------------------------------------------------------------------
// fp32 -> bf16 conversion (8 elements / thread, vectorized)
// ---------------------------------------------------------------------------
__global__ __launch_bounds__(256)
void cvt_f32_to_bf16(const float* __restrict__ in, __bf16* __restrict__ out, int n) {
    int i = (blockIdx.x * blockDim.x + threadIdx.x) * 8;
    if (i < n) {
        float4 f0 = *(const float4*)(in + i);
        float4 f1 = *(const float4*)(in + i + 4);
        v8bf o;
        o[0] = (__bf16)f0.x; o[1] = (__bf16)f0.y;
        o[2] = (__bf16)f0.z; o[3] = (__bf16)f0.w;
        o[4] = (__bf16)f1.x; o[5] = (__bf16)f1.y;
        o[6] = (__bf16)f1.z; o[7] = (__bf16)f1.w;
        *(v8bf*)(out + i) = o;
    }
}

// ---------------------------------------------------------------------------
// C[M,N] = A[M,K] * W[N,K]^T   (bf16 inputs, fp32 accumulate/output)
// Wave tile: 32(M) x 64(N): two A operands share each B operand ->
// 8 v_wmma_f32_16x16x32_bf16 per K-step from 12 b128 loads (1.5 loads/WMMA).
// Block: 256 threads = 8 waves stacked along M -> 256(M) x 64(N) block tile.
// Grid: (N/64, M/256).
// ---------------------------------------------------------------------------
__global__ __launch_bounds__(256)
void gemm_bf16_wmma(const __bf16* __restrict__ A, const __bf16* __restrict__ W,
                    float* __restrict__ C, int M, int N, int K) {
    const int lane = threadIdx.x & 31;
    const int wave = threadIdx.x >> 5;               // 0..7
    const int l16  = lane & 15;
    const int hi   = lane >> 4;                      // 0 or 1 (half-wave)
    const int m0   = blockIdx.y * 256 + wave * 32;
    const int n0   = blockIdx.x * 64;

    const __bf16* arow0 = A + (size_t)(m0 + l16) * K;        // rows m0..m0+15
    const __bf16* arow1 = A + (size_t)(m0 + 16 + l16) * K;   // rows m0+16..m0+31

    v8f acc0[4], acc1[4];
    #pragma unroll
    for (int t = 0; t < 4; ++t)
        #pragma unroll
        for (int e = 0; e < 8; ++e) { acc0[t][e] = 0.0f; acc1[t][e] = 0.0f; }

    for (int kk = 0; kk < K; kk += 32) {
        // A operands (16x32 bf16 each): lane<16 holds K{kk..kk+7, kk+16..kk+23},
        // lane>=16 holds K{kk+8..kk+15, kk+24..kk+31} of row M = m0(+16) + l16.
        v8bf a0lo = *(const v8bf*)(arow0 + kk + hi * 8);
        v8bf a0hi = *(const v8bf*)(arow0 + kk + 16 + hi * 8);
        v8bf a1lo = *(const v8bf*)(arow1 + kk + hi * 8);
        v8bf a1hi = *(const v8bf*)(arow1 + kk + 16 + hi * 8);
        v16bf a0, a1;
        #pragma unroll
        for (int e = 0; e < 8; ++e) {
            a0[e] = a0lo[e]; a0[e + 8] = a0hi[e];
            a1[e] = a1lo[e]; a1[e + 8] = a1hi[e];
        }

        #pragma unroll
        for (int t = 0; t < 4; ++t) {
            // B operand (32x16 bf16): lane = column N, same K order as A
            // (lanes 0-15: K kk..kk+7, kk+16..kk+23; lanes 16-31: K kk+8..kk+15, kk+24..kk+31).
            const __bf16* brow = W + (size_t)(n0 + t * 16 + l16) * K + kk + hi * 8;
            v8bf blo = *(const v8bf*)brow;
            v8bf bhi = *(const v8bf*)(brow + 16);
            v16bf b;
            #pragma unroll
            for (int e = 0; e < 8; ++e) { b[e] = blo[e]; b[e + 8] = bhi[e]; }
            acc0[t] = __builtin_amdgcn_wmma_f32_16x16x32_bf16(
                false, a0, false, b, (short)0, acc0[t], false, false);
            acc1[t] = __builtin_amdgcn_wmma_f32_16x16x32_bf16(
                false, a1, false, b, (short)0, acc1[t], false, false);
        }
    }

    // C layout: VGPR r -> M = base + r + 8*hi, N = n0 + t*16 + l16
    #pragma unroll
    for (int t = 0; t < 4; ++t) {
        const int col = n0 + t * 16 + l16;
        #pragma unroll
        for (int r = 0; r < 8; ++r) {
            C[(size_t)(m0 + r + hi * 8) * N + col]      = acc0[t][r];
            C[(size_t)(m0 + 16 + r + hi * 8) * N + col] = acc1[t][r];
        }
    }
}

// ---------------------------------------------------------------------------
// Recurrent scan: 4096 independent 8x8 states, 8 lanes per state.
// Lane i owns row i of S. Flat lane id == (b*NN + n*BLK + i), so all
// global accesses are fully coalesced. Per step: 3 butterfly shuffles for
// ||k||^2 plus 16 width-8 broadcasts of k and q; everything else lane-local.
// ---------------------------------------------------------------------------
__global__ __launch_bounds__(256)
void delta_scan(const float* __restrict__ kp, const float* __restrict__ vp,
                const float* __restrict__ qp, float* __restrict__ out) {
    const int tid = blockIdx.x * blockDim.x + threadIdx.x;   // 0..32767

    float S[8];
    #pragma unroll
    for (int j = 0; j < 8; ++j) S[j] = 0.0f;

    for (int t = 0; t < TT; ++t) {
        const int off = t * BNFLAT + tid;
        const float kv = kp[off];
        const float vv = vp[off];
        const float qv = qp[off];

        // prefetch a few steps ahead (gfx1250 global_prefetch_b8)
        if (t + 4 < TT) {
            __builtin_prefetch(kp + off + 4 * BNFLAT, 0, 0);
            __builtin_prefetch(vp + off + 4 * BNFLAT, 0, 0);
            __builtin_prefetch(qp + off + 4 * BNFLAT, 0, 0);
        }

        // ||k_t|| over the 8-lane group
        float s2 = kv * kv;
        s2 += __shfl_xor(s2, 1, 8);
        s2 += __shfl_xor(s2, 2, 8);
        s2 += __shfl_xor(s2, 4, 8);
        const float inv = 1.0f / (sqrtf(s2) + 1e-6f);

        // broadcast k, compute retrieved_i = sum_j S[i][j] * k_norm[j]
        float kn[8];
        float retr = 0.0f;
        #pragma unroll
        for (int j = 0; j < 8; ++j) {
            kn[j] = __shfl(kv, j, 8) * inv;
            retr = fmaf(S[j], kn[j], retr);
        }
        const float delta = vv - retr;

        // S_new = tanh(S + outer(delta, k_norm)); Sq_i = sum_j S_new[i][j]*q[j]
        float sq = 0.0f;
        #pragma unroll
        for (int j = 0; j < 8; ++j) {
            S[j] = tanhf(fmaf(delta, kn[j], S[j]));
            sq = fmaf(S[j], __shfl(qv, j, 8), sq);
        }

        // out = Sq * silu(Sq) = Sq * Sq * sigmoid(Sq)
        const float sig = 1.0f / (1.0f + __expf(-sq));
        out[off] = sq * sq * sig;
    }
}

// ---------------------------------------------------------------------------
// Host-side launcher
// ---------------------------------------------------------------------------
extern "C" void kernel_launch(void* const* d_in, const int* in_sizes, int n_in,
                              void* d_out, int out_size, void* d_ws, size_t ws_size,
                              hipStream_t stream) {
    (void)in_sizes; (void)n_in; (void)out_size; (void)ws_size;

    const float* x  = (const float*)d_in[0];
    const float* Wk = (const float*)d_in[1];
    const float* Wv = (const float*)d_in[2];
    const float* Wq = (const float*)d_in[3];
    float* out = (float*)d_out;

    // Workspace carve-up
    char* ws = (char*)d_ws;
    __bf16* xb  = (__bf16*)ws;  ws += (size_t)MROWS * DD * sizeof(__bf16);   // 64 MB
    __bf16* wkb = (__bf16*)ws;  ws += (size_t)NN * DD * sizeof(__bf16);      //  2 MB
    __bf16* wvb = (__bf16*)ws;  ws += (size_t)NN * DD * sizeof(__bf16);
    __bf16* wqb = (__bf16*)ws;  ws += (size_t)NN * DD * sizeof(__bf16);
    float*  kb  = (float*)ws;   ws += (size_t)MROWS * NN * sizeof(float);    // 128 MB
    float*  vb  = (float*)ws;   ws += (size_t)MROWS * NN * sizeof(float);
    float*  qb  = (float*)ws;   ws += (size_t)MROWS * NN * sizeof(float);

    // 1) fp32 -> bf16 conversions (x stays resident in the 192 MB L2 after this)
    const int nx = MROWS * DD;                 // 33,554,432
    const int nw = NN * DD;                    //  1,048,576
    cvt_f32_to_bf16<<<nx / (256 * 8), 256, 0, stream>>>(x,  xb,  nx);
    cvt_f32_to_bf16<<<nw / (256 * 8), 256, 0, stream>>>(Wk, wkb, nw);
    cvt_f32_to_bf16<<<nw / (256 * 8), 256, 0, stream>>>(Wv, wvb, nw);
    cvt_f32_to_bf16<<<nw / (256 * 8), 256, 0, stream>>>(Wq, wqb, nw);

    // 2) three WMMA projection GEMMs: k/v/q = x @ W^T
    dim3 ggrid(NN / 64, MROWS / 256);          // (16, 128)
    gemm_bf16_wmma<<<ggrid, 256, 0, stream>>>(xb, wkb, kb, MROWS, NN, DD);
    gemm_bf16_wmma<<<ggrid, 256, 0, stream>>>(xb, wvb, vb, MROWS, NN, DD);
    gemm_bf16_wmma<<<ggrid, 256, 0, stream>>>(xb, wqb, qb, MROWS, NN, DD);

    // 3) sequential delta-rule scan over T
    delta_scan<<<BNFLAT / 256, 256, 0, stream>>>(kb, vb, qb, out);
}
